// NonLocal_6433861010119
// MI455X (gfx1250) — matrix operand order repaired and checked by reading, not compiled, not run on hardware
//
#include <hip/hip_runtime.h>
#include <hip/hip_bf16.h>

// ---------------------------------------------------------------------------
// NonLocal attention block for MI455X (gfx1250, wave32, WMMA bf16 16x16x32)
//   b=16, c=512, h=w=32 (n=m=1024), 32 groups
// Projection GEMMs stage their workgroup-shared B panel through LDS
// (double-buffered, async global->LDS DMA via global_load_async_to_lds_b128),
// since all 8 waves consume the identical panel. Attention is fused
// flash-style with a parallel online-softmax epilogue.
// ---------------------------------------------------------------------------

typedef __bf16 bf16;
typedef bf16  v16bf __attribute__((ext_vector_type(16)));
typedef float v8f   __attribute__((ext_vector_type(8)));

#define B_   16
#define C_   512
#define N_   1024
#define GROUPS_ 32

#if __has_builtin(__builtin_amdgcn_global_load_async_to_lds_b128) && \
    __has_builtin(__builtin_amdgcn_s_wait_asynccnt)
#define ASYNC_LDS 1
typedef int v4i __attribute__((__vector_size__(16)));
typedef __attribute__((address_space(1))) v4i gv4i;   // global int4
typedef __attribute__((address_space(3))) v4i lv4i;   // LDS int4
__device__ __forceinline__ void async_cp16(const bf16* g, bf16* l) {
  __builtin_amdgcn_global_load_async_to_lds_b128((gv4i*)g, (lv4i*)l, 0, 0);
}
#else
#define ASYNC_LDS 0
#endif

__device__ __forceinline__ v8f wmma_bf16(v16bf a, v16bf b, v8f c) {
  // (neg_a, A, neg_b, B, c_mod, C, reuse_a, reuse_b)
  return __builtin_amdgcn_wmma_f32_16x16x32_bf16(false, a, false, b, (short)0, c,
                                                 false, false);
}

// A-matrix 16x32 bf16 fragment: lane = (lo,hi); row M = lo;
// lane holds K = hi*16 .. hi*16+15 contiguous (matches load_b_frag).
__device__ __forceinline__ v16bf load_a_frag(const bf16* row, int hi) {
  union { v16bf v; uint4 q[2]; } u;
  u.q[0] = *(const uint4*)(row + hi * 16);
  u.q[1] = *(const uint4*)(row + hi * 16 + 8);
  return u.v;
}

// B-matrix 32x16 bf16 fragment: lane = (lo,hi); col N = lo; lane holds
// K = hi*16 .. hi*16+15 contiguous. `col` must already include the hi*16 K
// offset.
__device__ __forceinline__ v16bf load_b_frag(const bf16* col) {
  union { v16bf v; uint4 q[2]; } u;
  u.q[0] = *(const uint4*)(col);
  u.q[1] = *(const uint4*)(col + 8);
  return u.v;
}

// ---------------------------------------------------------------------------
// Shared GEMM mainloop for the projection kernels.
//   acc[t] (16x16, t over 4 col-tiles) += A[16 rows x 512] * B[512 x 64]
// A rows are wave-private (global). The 64-column B panel is identical for
// all 8 waves -> staged through LDS, double-buffered, 32 k-values per step.
// Bsm rows padded to 40 elements to spread LDS banks.
// ---------------------------------------------------------------------------
__device__ __forceinline__ void gemm_mainloop_16x64(
    const bf16* __restrict__ arow,        // + lane row offset already applied
    const bf16* __restrict__ bbase,       // column c of B panel at bbase + c*C_
    bf16 (&Bsm)[2][64][40], int tid, int lo, int hi, v8f (&acc)[4]) {
  const int col = tid >> 2;               // 0..63
  const int seg = (tid & 3) * 8;          // 0,8,16,24 (bf16 elems, 16B chunks)
  const bf16* bsrc = bbase + (size_t)col * C_ + seg;

  // prologue: stage k-step 0 into buffer 0
#if ASYNC_LDS
  async_cp16(bsrc, &Bsm[0][col][seg]);
  __builtin_amdgcn_s_wait_asynccnt(0);
#else
  *(uint4*)&Bsm[0][col][seg] = *(const uint4*)bsrc;
#endif
  __syncthreads();

  v16bf af = load_a_frag(arow, hi);

#pragma unroll
  for (int ks = 0; ks < 16; ++ks) {
    const int cur = ks & 1;
#if !ASYNC_LDS
    uint4 stg;
#endif
    if (ks < 15) {   // issue stage of next panel before consuming current one
#if ASYNC_LDS
      async_cp16(bsrc + (ks + 1) * 32, &Bsm[cur ^ 1][col][seg]);
#else
      stg = *(const uint4*)(bsrc + (ks + 1) * 32);
#endif
    }
    v16bf afn;
    if (ks < 15) afn = load_a_frag(arow + (ks + 1) * 32, hi);

#pragma unroll
    for (int t = 0; t < 4; ++t) {
      v16bf bfg = load_b_frag(&Bsm[cur][t * 16 + lo][hi * 16]);
      acc[t] = wmma_bf16(af, bfg, acc[t]);
    }
    if (ks < 15) {
      af = afn;
#if ASYNC_LDS
      __builtin_amdgcn_s_wait_asynccnt(0);
#else
      *(uint4*)&Bsm[cur ^ 1][col][seg] = stg;
#endif
      __syncthreads();
    }
  }
}

// ---------------------------------------------------------------------------
// GroupNorm: inp (b,c,n) fp32 -> xnt (b,n,c) bf16  (position-major for WMMA)
// ---------------------------------------------------------------------------
__global__ __launch_bounds__(256) void gn_kernel(const float* __restrict__ inp,
                                                 const float* __restrict__ gamma,
                                                 const float* __restrict__ beta,
                                                 bf16* __restrict__ xnt) {
  const int g = blockIdx.x, b = blockIdx.y, tid = threadIdx.x;
  const size_t bbase = (size_t)b * C_ * N_;
  const float* gbase = inp + bbase + (size_t)g * 16 * N_;  // group slab contiguous

  float s = 0.f, s2 = 0.f;
  for (int i = tid; i < 16 * N_; i += 256) {
    float v = gbase[i];
    s += v; s2 += v * v;
  }
  __shared__ float rs[256], rs2[256];
  rs[tid] = s; rs2[tid] = s2;
  __syncthreads();
  for (int st = 128; st > 0; st >>= 1) {
    if (tid < st) { rs[tid] += rs[tid + st]; rs2[tid] += rs2[tid + st]; }
    __syncthreads();
  }
  const float mean = rs[0] * (1.f / 16384.f);
  const float var  = rs2[0] * (1.f / 16384.f) - mean * mean;
  const float inv  = rsqrtf(var + 1e-6f);

  for (int i = tid; i < 16 * N_; i += 256) {
    int ci = i & 15, pos = i >> 4;
    int c  = g * 16 + ci;
    float v = inp[bbase + (size_t)c * N_ + pos];
    float y = (v - mean) * inv * gamma[c] + beta[c];
    xnt[((size_t)b * N_ + pos) * C_ + c] = (bf16)y;
  }
}

// ---------------------------------------------------------------------------
// fp32 -> bf16 weight conversion
// ---------------------------------------------------------------------------
__global__ __launch_bounds__(256) void cvt_kernel(const float* __restrict__ src,
                                                  bf16* __restrict__ dst, int n) {
  int i = blockIdx.x * 256 + threadIdx.x;
  if (i < n) dst[i] = (bf16)src[i];
}

// ---------------------------------------------------------------------------
// Q/K projection: Out(b,n,c) = scale * (W @ xnt^T + bias)
// A = W rows (c_out), B = xnt columns (n, LDS-staged), D(c_out, n).
// ---------------------------------------------------------------------------
__global__ __launch_bounds__(256) void proj_qk_kernel(const bf16* __restrict__ xnt,
                                                      const bf16* __restrict__ Wb,
                                                      const float* __restrict__ bias,
                                                      bf16* __restrict__ Out,
                                                      float scale) {
  const int tid = threadIdx.x, lane = tid & 31, w = tid >> 5;
  const int lo = lane & 15, hi = lane >> 4;
  const int n0 = blockIdx.x * 64;
  const int crow = blockIdx.y * 128 + w * 16;
  const int b = blockIdx.z;
  const bf16* xb = xnt + (size_t)b * N_ * C_;

  __shared__ alignas(16) bf16 Bsm[2][64][40];

  const v8f vz = {0.f,0.f,0.f,0.f,0.f,0.f,0.f,0.f};
  v8f acc[4] = {vz, vz, vz, vz};

  gemm_mainloop_16x64(Wb + (size_t)(crow + lo) * C_, xb + (size_t)n0 * C_,
                      Bsm, tid, lo, hi, acc);

  // bias along M rows: c = crow + hi*8 + r
  const float* bp = bias + crow + hi * 8;
  float4 b0 = *(const float4*)bp;
  float4 b1 = *(const float4*)(bp + 4);
  float bb[8] = {b0.x, b0.y, b0.z, b0.w, b1.x, b1.y, b1.z, b1.w};

#pragma unroll
  for (int nt = 0; nt < 4; ++nt) {
    union { uint4 q; bf16 h[8]; } st;
#pragma unroll
    for (int r = 0; r < 8; ++r) st.h[r] = (bf16)((acc[nt][r] + bb[r]) * scale);
    *(uint4*)(Out + ((size_t)b * N_ + n0 + nt * 16 + lo) * C_ + crow + hi * 8) = st.q;
  }
}

// ---------------------------------------------------------------------------
// V projection (operands swapped): D(m, c_out) = xnt @ Wv^T, store (b,c,m)
// B = Wv rows (c_out, LDS-staged, shared by all waves).
// ---------------------------------------------------------------------------
__global__ __launch_bounds__(256) void proj_v_kernel(const bf16* __restrict__ xnt,
                                                     const bf16* __restrict__ Wb,
                                                     const float* __restrict__ bias,
                                                     bf16* __restrict__ Vb) {
  const int tid = threadIdx.x, lane = tid & 31, w = tid >> 5;
  const int lo = lane & 15, hi = lane >> 4;
  const int c0 = blockIdx.x * 64;
  const int mrow = blockIdx.y * 128 + w * 16;
  const int b = blockIdx.z;
  const bf16* xb = xnt + (size_t)b * N_ * C_;

  __shared__ alignas(16) bf16 Bsm[2][64][40];

  const v8f vz = {0.f,0.f,0.f,0.f,0.f,0.f,0.f,0.f};
  v8f acc[4] = {vz, vz, vz, vz};

  gemm_mainloop_16x64(xb + (size_t)(mrow + lo) * C_, Wb + (size_t)c0 * C_,
                      Bsm, tid, lo, hi, acc);

#pragma unroll
  for (int t = 0; t < 4; ++t) {
    const int cc = c0 + t * 16 + lo;          // N (fixed per lane)
    const float bv = bias[cc];
    union { uint4 q; bf16 h[8]; } st;
#pragma unroll
    for (int r = 0; r < 8; ++r) st.h[r] = (bf16)(acc[t][r] + bv);
    *(uint4*)(Vb + ((size_t)b * C_ + cc) * N_ + mrow + hi * 8) = st.q;
  }
}

// ---------------------------------------------------------------------------
// Fused attention: per (batch, 16-row n-block). 8 waves x 64-wide c-slices.
// Online softmax over m in 64-wide steps. Output A(c,n) stored as (b,n,c) bf16.
// ---------------------------------------------------------------------------
__global__ __launch_bounds__(256) void attn_kernel(const bf16* __restrict__ Qt,
                                                   const bf16* __restrict__ Kt,
                                                   const bf16* __restrict__ Vb,
                                                   bf16* __restrict__ Abuf) {
  const int tid = threadIdx.x, lane = tid & 31, w = tid >> 5;
  const int lo = lane & 15, hi = lane >> 4;
  const int n0 = blockIdx.x * 16;
  const int b = blockIdx.y;
  const int cw = w * 64;                       // this wave's c-slice

  __shared__ alignas(16) float Ssm[16][64];
  __shared__ alignas(16) bf16  Psm[16][64];
  __shared__ alignas(16) float red[16][16];    // tree-reduce scratch
  __shared__ float s_runmax[16], s_runsum[16], s_corr[16], s_mnew[16];

  const bf16* Qb = Qt + (size_t)b * N_ * C_;
  const bf16* Kb = Kt + (size_t)b * N_ * C_;
  const bf16* Vbase = Vb + (size_t)b * C_ * N_;

  if (tid < 16) { s_runmax[tid] = -__builtin_huge_valf(); s_runsum[tid] = 0.f; }

  // Q A-fragments are m-invariant: hoist out of the loop (16 VGPRs).
  v16bf aQ[2];
  aQ[0] = load_a_frag(Qb + (size_t)(n0 + lo) * C_ + cw, hi);
  aQ[1] = load_a_frag(Qb + (size_t)(n0 + lo) * C_ + cw + 32, hi);

  const v8f vz = {0.f,0.f,0.f,0.f,0.f,0.f,0.f,0.f};
  v8f oacc[4] = {vz, vz, vz, vz};
  __syncthreads();

  for (int ms = 0; ms < 16; ++ms) {
    const int m0 = ms * 64;
    // zero the S reduction tile
    ((float4*)&Ssm[0][0])[tid] = make_float4(0.f, 0.f, 0.f, 0.f);
    __syncthreads();

    // ---- partial S = Q^T K over this wave's c-slice (pipelined B loads) ----
    const bf16* kcol[4];
#pragma unroll
    for (int mt = 0; mt < 4; ++mt)
      kcol[mt] = Kb + (size_t)(m0 + mt * 16 + lo) * C_ + cw + hi * 16;

    v8f sacc[4] = {vz, vz, vz, vz};
    v16bf bcur = load_b_frag(kcol[0]);
#pragma unroll
    for (int ks = 0; ks < 2; ++ks) {
#pragma unroll
      for (int mt = 0; mt < 4; ++mt) {
        const int nid = ks * 4 + mt + 1;
        v16bf bnxt;
        if (nid < 8) bnxt = load_b_frag(kcol[nid & 3] + (nid >> 2) * 32);
        sacc[mt] = wmma_bf16(aQ[ks], bcur, sacc[mt]);
        if (nid < 8) bcur = bnxt;
      }
    }
    if (ms < 15)   // prefetch next K tile
      __builtin_prefetch(Kb + (size_t)(m0 + 64 + lane) * C_ + cw, 0, 1);

    // reduce partial S across the 8 c-slices
#pragma unroll
    for (int mt = 0; mt < 4; ++mt)
#pragma unroll
      for (int r = 0; r < 8; ++r)
        atomicAdd(&Ssm[hi * 8 + r][mt * 16 + lo], sacc[mt][r]);
    __syncthreads();

    // ---- parallel online softmax ----
    // phase A: per-thread max over 4 elements
    {
      const int r = tid >> 4, j0 = (tid & 15) * 4;
      float4 sv = *(const float4*)&Ssm[r][j0];
      red[r][tid & 15] = fmaxf(fmaxf(sv.x, sv.y), fmaxf(sv.z, sv.w));
    }
    __syncthreads();
    // phase B: row max + correction factor (16 threads)
    if (tid < 16) {
      float cmax = red[tid][0];
#pragma unroll
      for (int j = 1; j < 16; ++j) cmax = fmaxf(cmax, red[tid][j]);
      float mold = s_runmax[tid];
      float mnew = fmaxf(mold, cmax);
      s_corr[tid] = __expf(mold - mnew);
      s_mnew[tid] = mnew;
      s_runmax[tid] = mnew;
    }
    __syncthreads();
    // phase C: exp + P write + partial sums (all 256 threads, 4 elems each)
    {
      const int r = tid >> 4, j0 = (tid & 15) * 4;
      const float mnew = s_mnew[r];
      float4 sv = *(const float4*)&Ssm[r][j0];
      float p0 = __expf(sv.x - mnew), p1 = __expf(sv.y - mnew);
      float p2 = __expf(sv.z - mnew), p3 = __expf(sv.w - mnew);
      union { uint2 q; bf16 h[4]; } st;
      st.h[0] = (bf16)p0; st.h[1] = (bf16)p1; st.h[2] = (bf16)p2; st.h[3] = (bf16)p3;
      *(uint2*)&Psm[r][j0] = st.q;
      red[r][tid & 15] = (p0 + p1) + (p2 + p3);
    }
    __syncthreads();
    // phase D: running-sum update (16 threads; overlaps with AV below)
    if (tid < 16) {
      float cs = red[tid][0];
#pragma unroll
      for (int j = 1; j < 16; ++j) cs += red[tid][j];
      s_runsum[tid] = s_runsum[tid] * s_corr[tid] + cs;
    }

    // ---- rescale accumulator and accumulate V @ P^T (pipelined A loads) ----
    const float cf = s_corr[lo];
#pragma unroll
    for (int t = 0; t < 4; ++t) oacc[t] *= cf;

    const bf16* vrow[4];
#pragma unroll
    for (int t = 0; t < 4; ++t)
      vrow[t] = Vbase + (size_t)(cw + t * 16 + lo) * N_ + m0;

    v16bf av = load_a_frag(vrow[0], hi);
#pragma unroll
    for (int ks = 0; ks < 2; ++ks) {
      v16bf pb = load_b_frag(&Psm[lo][ks * 32 + hi * 16]);   // LDS
#pragma unroll
      for (int t = 0; t < 4; ++t) {
        const int nid = ks * 4 + t + 1;
        v16bf avn;
        if (nid < 8) avn = load_a_frag(vrow[nid & 3] + (nid >> 2) * 32, hi);
        oacc[t] = wmma_bf16(av, pb, oacc[t]);
        if (nid < 8) av = avn;
      }
    }
    __syncthreads();   // protect Ssm/Psm/red before next step
  }

  const float fac = 1.0f / s_runsum[lo];
#pragma unroll
  for (int t = 0; t < 4; ++t) {
    union { uint4 q; bf16 h[8]; } st;
#pragma unroll
    for (int r = 0; r < 8; ++r) st.h[r] = (bf16)(oacc[t][r] * fac);
    *(uint4*)(Abuf + ((size_t)b * N_ + n0 + lo) * C_ + cw + t * 16 + hi * 8) = st.q;
  }
}

// ---------------------------------------------------------------------------
// Output projection + residual: out(b,c,n) = inp + Wo @ A + bo
// Computed transposed: D(n, c_out) = Abuf @ Wo^T, B = Wo rows (LDS-staged).
// ---------------------------------------------------------------------------
__global__ __launch_bounds__(256) void proj_o_kernel(const bf16* __restrict__ Abuf,
                                                     const bf16* __restrict__ Wb,
                                                     const float* __restrict__ bo,
                                                     const float* __restrict__ inp,
                                                     float* __restrict__ out) {
  const int tid = threadIdx.x, lane = tid & 31, w = tid >> 5;
  const int lo = lane & 15, hi = lane >> 4;
  const int c0 = blockIdx.x * 64;
  const int nrow = blockIdx.y * 128 + w * 16;
  const int b = blockIdx.z;
  const bf16* Ab = Abuf + (size_t)b * N_ * C_;

  __shared__ alignas(16) bf16 Bsm[2][64][40];

  const v8f vz = {0.f,0.f,0.f,0.f,0.f,0.f,0.f,0.f};
  v8f acc[4] = {vz, vz, vz, vz};

  gemm_mainloop_16x64(Ab + (size_t)(nrow + lo) * C_, Wb + (size_t)c0 * C_,
                      Bsm, tid, lo, hi, acc);

#pragma unroll
  for (int t = 0; t < 4; ++t) {
    const int cc = c0 + t * 16 + lo;                  // N (fixed per lane)
    const float bb = bo[cc];
    const size_t base = ((size_t)b * C_ + cc) * N_ + nrow + hi * 8;
    float4 r0 = *(const float4*)(inp + base);
    float4 r1 = *(const float4*)(inp + base + 4);
    float4 o0 = make_float4(r0.x + acc[t][0] + bb, r0.y + acc[t][1] + bb,
                            r0.z + acc[t][2] + bb, r0.w + acc[t][3] + bb);
    float4 o1 = make_float4(r1.x + acc[t][4] + bb, r1.y + acc[t][5] + bb,
                            r1.z + acc[t][6] + bb, r1.w + acc[t][7] + bb);
    *(float4*)(out + base)     = o0;
    *(float4*)(out + base + 4) = o1;
  }
}

// ---------------------------------------------------------------------------
// Host launcher. Workspace layout (bytes):
//   [0,16M)   xnt (b,n,c) bf16  -- reused as Abuf by attention
//   [16M,32M) Qt  (b,n,c) bf16  (pre-scaled by 1/sqrt(c))
//   [32M,48M) Kt  (b,m,c) bf16
//   [48M,64M) Vb  (b,c,m) bf16
//   [64M,66M) Wq/Wk/Wv/Wo bf16 copies (512KB each)
// ---------------------------------------------------------------------------
extern "C" void kernel_launch(void* const* d_in, const int* in_sizes, int n_in,
                              void* d_out, int out_size, void* d_ws, size_t ws_size,
                              hipStream_t stream) {
  const float* inp   = (const float*)d_in[0];
  const float* gamma = (const float*)d_in[1];
  const float* beta  = (const float*)d_in[2];
  const float* Wq    = (const float*)d_in[3];
  const float* bq    = (const float*)d_in[4];
  const float* Wk    = (const float*)d_in[5];
  const float* bk    = (const float*)d_in[6];
  const float* Wv    = (const float*)d_in[7];
  const float* bv    = (const float*)d_in[8];
  const float* Wo    = (const float*)d_in[9];
  const float* bo    = (const float*)d_in[10];
  float* out = (float*)d_out;

  char* ws = (char*)d_ws;
  bf16* xnt = (bf16*)(ws);
  bf16* Qt  = (bf16*)(ws + (size_t)(16u << 20));
  bf16* Kt  = (bf16*)(ws + (size_t)(32u << 20));
  bf16* Vb  = (bf16*)(ws + (size_t)(48u << 20));
  bf16* Wqb = (bf16*)(ws + (size_t)(64u << 20));
  bf16* Wkb = Wqb + C_ * C_;
  bf16* Wvb = Wkb + C_ * C_;
  bf16* Wob = Wvb + C_ * C_;
  bf16* Abuf = xnt;  // alias: xnt is dead once projections finish

  const int wn = C_ * C_;  // 262144

  gn_kernel<<<dim3(GROUPS_, B_), 256, 0, stream>>>(inp, gamma, beta, xnt);

  cvt_kernel<<<(wn + 255) / 256, 256, 0, stream>>>(Wq, Wqb, wn);
  cvt_kernel<<<(wn + 255) / 256, 256, 0, stream>>>(Wk, Wkb, wn);
  cvt_kernel<<<(wn + 255) / 256, 256, 0, stream>>>(Wv, Wvb, wn);
  cvt_kernel<<<(wn + 255) / 256, 256, 0, stream>>>(Wo, Wob, wn);

  const float qscale = 0.044194173824159216f;  // 1/sqrt(512)
  proj_qk_kernel<<<dim3(16, 4, B_), 256, 0, stream>>>(xnt, Wqb, bq, Qt, qscale);
  proj_qk_kernel<<<dim3(16, 4, B_), 256, 0, stream>>>(xnt, Wkb, bk, Kt, 1.0f);
  proj_v_kernel<<<dim3(8, 8, B_), 256, 0, stream>>>(xnt, Wvb, bv, Vb);

  attn_kernel<<<dim3(N_ / 16, B_), 256, 0, stream>>>(Qt, Kt, Vb, Abuf);

  proj_o_kernel<<<dim3(8, 8, B_), 256, 0, stream>>>(Abuf, Wob, bo, inp, out);
}
